// Transformer_12936441496192
// MI455X (gfx1250) — compile-verified, matrix operand order checked by reading, not run on hardware
//
#include <hip/hip_runtime.h>

// ---------------- types for gfx1250 WMMA / TDM ----------------
typedef __bf16 bf16;
typedef __attribute__((ext_vector_type(16))) __bf16 v16bf;
typedef __attribute__((ext_vector_type(8)))  float  v8f;
typedef __attribute__((ext_vector_type(4)))  unsigned int u32x4;
typedef __attribute__((ext_vector_type(4)))  float  f32x4;
typedef __attribute__((ext_vector_type(8)))  int    i32x8;
typedef __attribute__((ext_vector_type(4)))  int    i32x4;

union FragB16 { v16bf v; u32x4 q[2]; bf16 e[16]; };

#define WMMA_BF16(A_, B_, C_) \
  __builtin_amdgcn_wmma_f32_16x16x32_bf16(false, (A_), false, (B_), (short)0, (C_), false, false)

#if defined(__has_builtin)
#if __has_builtin(__builtin_amdgcn_tensor_load_to_lds) && __has_builtin(__builtin_amdgcn_s_wait_tensorcnt)
#define HAVE_TDM 1
#endif
#endif

// ---------------- problem constants ----------------
constexpr int Lc = 12, Bc = 8, Sc = 1024, Dc = 768, Hc = 12, Fc = 3072;
constexpr int Mc = Bc * Sc;         // 8192 rows of activations
constexpr int DHc = Dc / Hc;        // 64

#ifdef HAVE_TDM
// ---------------------------------------------------------------------
// TDM: 2D tile load Global->LDS, 8-byte elements, optional LDS padding.
// Packs D# group0/group1 per CDNA5 ISA 8.3/8.4; groups 2/3 zero (2D).
//  lds_off   : LDS byte offset of tile start
//  gaddr     : global address of tile start
//  tile0     : tile width  in 8B units   (row bytes / 8)
//  tile1     : tile height in rows
//  stride0   : tensor row pitch in 8B units
//  pad_en/pi/pa : LDS padding (pi: pad every (1<<pi) 8B-units; pa: (pa+1) DWORDs)
// ---------------------------------------------------------------------
__device__ __forceinline__
void tdm_load_2d(unsigned lds_off, const void* gaddr,
                 unsigned tile0, unsigned tile1,
                 unsigned long long stride0,
                 unsigned pad_en, unsigned pad_int, unsigned pad_amt)
{
    unsigned long long ga = (unsigned long long)(uintptr_t)gaddr;
    u32x4 g0;
    g0[0] = 1u;                                         // count=1, user mode
    g0[1] = lds_off;                                    // lds_addr
    g0[2] = (unsigned)(ga & 0xffffffffu);               // global_addr[31:0]
    g0[3] = (unsigned)((ga >> 32) & 0x01ffffffu)        // global_addr[56:32]
          | (2u << 30);                                 // type=2 ("image")
    i32x8 g1;
    g1[0] = (int)((3u << 16)                            // data_size=8B, wg_mask=0
          | (pad_en << 20) | (pad_int << 22) | (pad_amt << 25));
    g1[1] = (int)((tile0 & 0xffffu) << 16);             // tensor_dim0[15:0] (== tile0)
    g1[2] = (int)(((tile0 >> 16) & 0xffffu)             // tensor_dim0[31:16]
          | ((tile1 & 0xffffu) << 16));                 // tensor_dim1[15:0] (== tile1)
    g1[3] = (int)(((tile1 >> 16) & 0xffffu)             // tensor_dim1[31:16]
          | ((tile0 & 0xffffu) << 16));                 // tile_dim0
    g1[4] = (int)(tile1 & 0xffffu);                     // tile_dim1 (tile_dim2=0)
    g1[5] = (int)(unsigned)(stride0 & 0xffffffffu);     // tensor_dim0_stride[31:0]
    g1[6] = (int)(unsigned)((stride0 >> 32) & 0xffffu); // stride[47:32] (dim1_stride=0)
    g1[7] = 0;
    i32x4 z4; z4[0] = 0; z4[1] = 0; z4[2] = 0; z4[3] = 0;
    i32x8 z8; for (int i = 0; i < 8; ++i) z8[i] = 0;
    // 6-arg form (clang-23 / therock-10.0): groups 0..3 + extra group + cpol
    __builtin_amdgcn_tensor_load_to_lds(g0, g1, z4, z4, z8, 0);
}
#endif

// =====================================================================
// LayerNorm: one block per row (768 elems), 256 threads, 3 elems/thread
// =====================================================================
template<int OUTBF>
__global__ __launch_bounds__(256)
void ln_kernel(const float* __restrict__ x, const float* __restrict__ g,
               const float* __restrict__ b, bf16* __restrict__ outB,
               float* __restrict__ outF)
{
    __shared__ float red[256];
    const int tid = threadIdx.x;
    const size_t row = blockIdx.x;
    const float* xr = x + row * Dc;

    float v0 = xr[tid], v1 = xr[tid + 256], v2 = xr[tid + 512];
    red[tid] = v0 + v1 + v2;
    __syncthreads();
    for (int off = 128; off; off >>= 1) {
        if (tid < off) red[tid] += red[tid + off];
        __syncthreads();
    }
    float mu = red[0] * (1.0f / (float)Dc);
    __syncthreads();

    float d0 = v0 - mu, d1 = v1 - mu, d2 = v2 - mu;
    red[tid] = d0 * d0 + d1 * d1 + d2 * d2;
    __syncthreads();
    for (int off = 128; off; off >>= 1) {
        if (tid < off) red[tid] += red[tid + off];
        __syncthreads();
    }
    float rstd = rsqrtf(red[0] * (1.0f / (float)Dc) + 1e-6f);

    float o0 = d0 * rstd * g[tid]       + b[tid];
    float o1 = d1 * rstd * g[tid + 256] + b[tid + 256];
    float o2 = d2 * rstd * g[tid + 512] + b[tid + 512];
    if (OUTBF) {
        bf16* o = outB + row * Dc;
        o[tid] = (bf16)o0; o[tid + 256] = (bf16)o1; o[tid + 512] = (bf16)o2;
    } else {
        float* o = outF + row * Dc;
        o[tid] = o0; o[tid + 256] = o1; o[tid + 512] = o2;
    }
}

// =====================================================================
// WMMA GEMM: C(MxN) = A(MxK,bf16) @ W(KxN,fp32->bf16) + bias, + epilogue
// Block tile 128x128x32; 8 wave32 waves as 4(M) x 2(N); A tile via TDM.
// =====================================================================
enum { EPI_BF16 = 0, EPI_RESID = 1, EPI_GELU = 2 };

template<int EPI>
__global__ __launch_bounds__(256)
void gemm_bf16_kernel(const bf16* __restrict__ A, const float* __restrict__ W,
                      const float* __restrict__ bias, const float* __restrict__ resid,
                      float* __restrict__ outF, bf16* __restrict__ outB,
                      int M, int N, int K)
{
    constexpr int BM = 128, BN = 128, BK = 32, BKP = 40; // 80B pitch (16B multiple)
    __shared__ bf16 sA[BM][BKP];   // [m][k]
    __shared__ bf16 sB[BN][BKP];   // [n][k]  (W transposed into LDS)

    const int tid   = threadIdx.x;
    const int wave  = tid >> 5;
    const int lane  = tid & 31;
    const int waveM = wave & 3;    // 0..3 -> 32 rows each
    const int waveN = wave >> 2;   // 0..1 -> 64 cols each
    const int lrow  = lane & 15;
    const int kh    = (lane >> 4) * 8;
    const int bm = blockIdx.y * BM;
    const int bn = blockIdx.x * BN;

    v8f acc[2][4];
    for (int mi = 0; mi < 2; ++mi)
        for (int ni = 0; ni < 4; ++ni)
            for (int e = 0; e < 8; ++e) acc[mi][ni][e] = 0.0f;

    const int nKt = K / BK;
    for (int kt = 0; kt < nKt; ++kt) {
        __syncthreads();   // prior-iteration LDS reads retired

        // ---- stage A tile: 128 rows x 64B, pitch 80B, via Tensor Data Mover ----
#ifdef HAVE_TDM
        if (wave == 0) {
            tdm_load_2d((unsigned)(uintptr_t)&sA[0][0],
                        A + (size_t)bm * K + (size_t)kt * BK,
                        /*tile0(8B)=*/BK * 2 / 8, /*tile1=*/BM,
                        /*stride0(8B)=*/(unsigned long long)K / 4,
                        /*pad_en=*/1, /*pad_int=*/3 /*64B*/, /*pad_amt=*/3 /*16B*/);
        }
#else
        #pragma unroll
        for (int i = 0; i < 2; ++i) {
            int idx = tid + i * 256;
            int r   = idx >> 2;
            int c8  = (idx & 3) * 8;
            u32x4 v = *(const u32x4*)(A + (size_t)(bm + r) * K + (size_t)kt * BK + c8);
            *(u32x4*)&sA[r][c8] = v;
        }
#endif
        // ---- stage W tile: 32x128 fp32 -> bf16, transposed into sB ----
        #pragma unroll
        for (int i = 0; i < 4; ++i) {
            int idx = tid + i * 256;
            int kk  = idx >> 5;            // 0..31
            int c4  = (idx & 31) * 4;      // 0..124
            f32x4 wv = *(const f32x4*)(W + (size_t)(kt * BK + kk) * N + bn + c4);
            sB[c4 + 0][kk] = (bf16)wv.x;
            sB[c4 + 1][kk] = (bf16)wv.y;
            sB[c4 + 2][kk] = (bf16)wv.z;
            sB[c4 + 3][kk] = (bf16)wv.w;
        }
        if (kt + 1 < nKt) { // global_prefetch_b8 next tiles
            __builtin_prefetch(A + (size_t)(bm + (tid >> 1)) * K + (size_t)(kt + 1) * BK, 0, 1);
            __builtin_prefetch(W + (size_t)((kt + 1) * BK + (tid >> 3)) * N + bn, 0, 1);
        }
#ifdef HAVE_TDM
        if (wave == 0) __builtin_amdgcn_s_wait_tensorcnt(0);
#endif
        __syncthreads();

        FragB16 af[2], bfr[4];
        #pragma unroll
        for (int mi = 0; mi < 2; ++mi) {
            int r = waveM * 32 + mi * 16 + lrow;
            af[mi].q[0] = *(const u32x4*)&sA[r][kh];
            af[mi].q[1] = *(const u32x4*)&sA[r][kh + 16];
        }
        #pragma unroll
        for (int ni = 0; ni < 4; ++ni) {
            int c = waveN * 64 + ni * 16 + lrow;
            bfr[ni].q[0] = *(const u32x4*)&sB[c][kh];
            bfr[ni].q[1] = *(const u32x4*)&sB[c][kh + 16];
        }
        #pragma unroll
        for (int mi = 0; mi < 2; ++mi)
            #pragma unroll
            for (int ni = 0; ni < 4; ++ni)
                acc[mi][ni] = WMMA_BF16(af[mi].v, bfr[ni].v, acc[mi][ni]);
    }

    // ---- epilogue
    #pragma unroll
    for (int mi = 0; mi < 2; ++mi) {
        #pragma unroll
        for (int ni = 0; ni < 4; ++ni) {
            int rbase = bm + waveM * 32 + mi * 16 + (lane >> 4) * 8;
            int col   = bn + waveN * 64 + ni * 16 + lrow;
            float bc = bias[col];
            #pragma unroll
            for (int j = 0; j < 8; ++j) {
                size_t off = (size_t)(rbase + j) * N + col;
                float v = acc[mi][ni][j] + bc;
                if (EPI == EPI_BF16) {
                    outB[off] = (bf16)v;
                } else if (EPI == EPI_RESID) {
                    outF[off] = resid[off] + v;
                } else { // exact GELU
                    float gl = 0.5f * v * (1.0f + erff(v * 0.70710678118654752f));
                    outB[off] = (bf16)gl;
                }
            }
        }
    }
}

// =====================================================================
// Flash-style attention: one block = 64 queries of one (b,h).
// Q/K tiles staged by TDM; V transposed manually; WMMA for QK^T and PV.
// =====================================================================
__global__ __launch_bounds__(256)
void attn_kernel(const bf16* __restrict__ Q, const bf16* __restrict__ Kb,
                 const bf16* __restrict__ V, const int* __restrict__ mask,
                 bf16* __restrict__ O)
{
    __shared__ bf16  sQ [64][72];      // [q][dh], pitch 144B
    __shared__ bf16  sKV[64][72];      // pass A: [key][dh]; pass C: [dh][key]
    __shared__ float sS [64][68];      // scores / probabilities
    __shared__ float red[64 * 4];
    __shared__ float rowM[64], rowL[64], rowA[64];
    __shared__ float smask[64];

    const int qb  = blockIdx.x;
    const int h   = blockIdx.y;
    const int b   = blockIdx.z;
    const int tid = threadIdx.x;
    const int wave = tid >> 5, lane = tid & 31;
    const int lrow = lane & 15, kh = (lane >> 4) * 8;
    const int mt = wave & 3;           // M tile (queries)
    const int ng = wave >> 2;          // N tile group (2 tiles)
    const int rq = tid >> 2;           // softmax row
    const int qq = tid & 3;            // column quarter

    const bf16* Qbase = Q  + ((size_t)b * Sc + (size_t)qb * 64) * Dc + h * DHc;
    const bf16* Kbase = Kb + (size_t)b * Sc * Dc + h * DHc;
    const bf16* Vbase = V  + (size_t)b * Sc * Dc + h * DHc;

    // ---- Q tile 64 x 128B, pitch 144B ----
#ifdef HAVE_TDM
    if (wave == 0) {
        tdm_load_2d((unsigned)(uintptr_t)&sQ[0][0], Qbase,
                    /*tile0=*/16, /*tile1=*/64,
                    /*stride0=*/(unsigned long long)Dc / 4,
                    /*pad_en=*/1, /*pad_int=*/4 /*128B*/, /*pad_amt=*/3 /*16B*/);
    }
#else
    #pragma unroll
    for (int i = 0; i < 2; ++i) {
        int idx = tid + i * 256;
        int r = idx >> 3, c = (idx & 7) * 8;
        *(u32x4*)&sQ[r][c] = *(const u32x4*)(Qbase + (size_t)r * Dc + c);
    }
#endif
    if (tid < 64) { rowM[tid] = -1e30f; rowL[tid] = 0.0f; }

    v8f oacc[2];
    for (int t = 0; t < 2; ++t)
        for (int e = 0; e < 8; ++e) oacc[t][e] = 0.0f;

    for (int kt = 0; kt < Sc / 64; ++kt) {
        __syncthreads();
        // ---- K tile straight in (B-operand of Q.K^T) ----
#ifdef HAVE_TDM
        if (wave == 0) {
            tdm_load_2d((unsigned)(uintptr_t)&sKV[0][0],
                        Kbase + (size_t)(kt * 64) * Dc,
                        /*tile0=*/16, /*tile1=*/64,
                        /*stride0=*/(unsigned long long)Dc / 4,
                        /*pad_en=*/1, /*pad_int=*/4, /*pad_amt=*/3);
        }
#else
        #pragma unroll
        for (int i = 0; i < 2; ++i) {
            int idx = tid + i * 256;
            int r = idx >> 3, c = (idx & 7) * 8;
            *(u32x4*)&sKV[r][c] = *(const u32x4*)(Kbase + (size_t)(kt * 64 + r) * Dc + c);
        }
#endif
        if (tid < 64) smask[tid] = (float)mask[(size_t)b * Sc + kt * 64 + tid];
#ifdef HAVE_TDM
        if (wave == 0) __builtin_amdgcn_s_wait_tensorcnt(0); // Q (first iter) + K
#endif
        __syncthreads();

        // ---- S tile = Q @ K^T (scale + mask applied on write) ----
        #pragma unroll
        for (int t = 0; t < 2; ++t) {
            int nt = ng * 2 + t;
            v8f sacc;
            for (int e = 0; e < 8; ++e) sacc[e] = 0.0f;
            FragB16 aF, bF;
            #pragma unroll
            for (int ks = 0; ks < 2; ++ks) {
                int kb = ks * 32;
                int r = mt * 16 + lrow;
                aF.q[0] = *(const u32x4*)&sQ[r][kb + kh];
                aF.q[1] = *(const u32x4*)&sQ[r][kb + kh + 16];
                int c = nt * 16 + lrow;
                bF.q[0] = *(const u32x4*)&sKV[c][kb + kh];
                bF.q[1] = *(const u32x4*)&sKV[c][kb + kh + 16];
                sacc = WMMA_BF16(aF.v, bF.v, sacc);
            }
            int colb = nt * 16 + lrow;
            int rb   = mt * 16 + (lane >> 4) * 8;
            float mterm = -10000.0f * (1.0f - smask[colb]);
            #pragma unroll
            for (int j = 0; j < 8; ++j)
                sS[rb + j][colb] = sacc[j] * 0.125f + mterm;
        }
        __syncthreads();

        // ---- online softmax: 4 threads per row ----
        {
            float pm = -1e30f;
            #pragma unroll
            for (int c = 0; c < 16; ++c) pm = fmaxf(pm, sS[rq][qq * 16 + c]);
            red[rq * 4 + qq] = pm;
        }
        __syncthreads();
        if (tid < 64) {
            float mnew = rowM[tid];
            for (int q2 = 0; q2 < 4; ++q2) mnew = fmaxf(mnew, red[tid * 4 + q2]);
            rowA[tid] = __expf(rowM[tid] - mnew);
            rowM[tid] = mnew;
        }
        __syncthreads();
        {
            float mnew = rowM[rq], ps = 0.0f;
            #pragma unroll
            for (int c = 0; c < 16; ++c) {
                float e = __expf(sS[rq][qq * 16 + c] - mnew);
                sS[rq][qq * 16 + c] = e;
                ps += e;
            }
            red[rq * 4 + qq] = ps;
        }
        __syncthreads();
        if (tid < 64) {
            float s4 = 0.0f;
            for (int q2 = 0; q2 < 4; ++q2) s4 += red[tid * 4 + q2];
            rowL[tid] = rowL[tid] * rowA[tid] + s4;
        }
        __syncthreads();

        // ---- V tile transposed: sKV[dh][key] ----
        #pragma unroll
        for (int i = 0; i < 2; ++i) {
            int idx = tid + i * 256;
            int key = idx >> 3, c = (idx & 7) * 8;
            u32x4 vv = *(const u32x4*)(Vbase + (size_t)(kt * 64 + key) * Dc + c);
            const bf16* pv = (const bf16*)&vv;
            #pragma unroll
            for (int j = 0; j < 8; ++j) sKV[c + j][key] = pv[j];
        }
        __syncthreads();

        // ---- O = O*alpha + P @ V ----
        #pragma unroll
        for (int t = 0; t < 2; ++t) {
            int nt = ng * 2 + t;
            int rb = mt * 16 + (lane >> 4) * 8;
            #pragma unroll
            for (int j = 0; j < 8; ++j) oacc[t][j] *= rowA[rb + j];
            #pragma unroll
            for (int ks = 0; ks < 2; ++ks) {
                int kb = ks * 32;
                FragB16 aF, bF;
                int r = mt * 16 + lrow;
                #pragma unroll
                for (int e = 0; e < 8; ++e) aF.e[e]     = (bf16)sS[r][kb + kh + e];
                #pragma unroll
                for (int e = 0; e < 8; ++e) aF.e[8 + e] = (bf16)sS[r][kb + kh + 16 + e];
                int c = nt * 16 + lrow;
                bF.q[0] = *(const u32x4*)&sKV[c][kb + kh];
                bF.q[1] = *(const u32x4*)&sKV[c][kb + kh + 16];
                oacc[t] = WMMA_BF16(aF.v, bF.v, oacc[t]);
            }
        }
    }
    __syncthreads();

    bf16* Ob = O + ((size_t)b * Sc + (size_t)qb * 64) * Dc + h * DHc;
    #pragma unroll
    for (int t = 0; t < 2; ++t) {
        int nt  = ng * 2 + t;
        int rb  = mt * 16 + (lane >> 4) * 8;
        int col = nt * 16 + lrow;
        #pragma unroll
        for (int j = 0; j < 8; ++j) {
            float val = oacc[t][j] / rowL[rb + j];
            Ob[(size_t)(rb + j) * Dc + col] = (bf16)val;
        }
    }
}

// =====================================================================
// Host-side orchestration
// =====================================================================
extern "C" void kernel_launch(void* const* d_in, const int* in_sizes, int n_in,
                              void* d_out, int out_size, void* d_ws, size_t ws_size,
                              hipStream_t stream)
{
    const float* x    = (const float*)d_in[0];
    const int*   mask = (const int*)  d_in[1];
    const float* Wq = (const float*)d_in[2];  const float* bq = (const float*)d_in[3];
    const float* Wk = (const float*)d_in[4];  const float* bk = (const float*)d_in[5];
    const float* Wv = (const float*)d_in[6];  const float* bv = (const float*)d_in[7];
    const float* Wo = (const float*)d_in[8];  const float* bo = (const float*)d_in[9];
    const float* g1 = (const float*)d_in[10]; const float* b1 = (const float*)d_in[11];
    const float* W1 = (const float*)d_in[12]; const float* c1 = (const float*)d_in[13];
    const float* W2 = (const float*)d_in[14]; const float* c2 = (const float*)d_in[15];
    const float* g2 = (const float*)d_in[16]; const float* b2 = (const float*)d_in[17];
    const float* gn = (const float*)d_in[18]; const float* bn = (const float*)d_in[19];

    // workspace carve-up (all 16B aligned)
    char* w = (char*)d_ws;
    bf16* hbuf = (bf16*)w;  w += (size_t)Mc * Dc * sizeof(bf16);
    bf16* qbuf = (bf16*)w;  w += (size_t)Mc * Dc * sizeof(bf16);
    bf16* kbuf = (bf16*)w;  w += (size_t)Mc * Dc * sizeof(bf16);
    bf16* vbuf = (bf16*)w;  w += (size_t)Mc * Dc * sizeof(bf16);
    bf16* obuf = (bf16*)w;  w += (size_t)Mc * Dc * sizeof(bf16);
    bf16* ubuf = (bf16*)w;  w += (size_t)Mc * Fc * sizeof(bf16);
    float* P0  = (float*)w; w += (size_t)Mc * Dc * sizeof(float);
    float* P1  = (float*)w; w += (size_t)Mc * Dc * sizeof(float);

    dim3 blk(256);
    dim3 gridD(Dc / 128, Mc / 128);   // (6, 64)
    dim3 gridF(Fc / 128, Mc / 128);   // (24, 64)
    dim3 gridA(Sc / 64, Hc, Bc);      // (16, 12, 8)

    const float* xin = x;
    for (int l = 0; l < Lc; ++l) {
        const float* wq = Wq + (size_t)l * Dc * Dc;  const float* bq_ = bq + (size_t)l * Dc;
        const float* wk = Wk + (size_t)l * Dc * Dc;  const float* bk_ = bk + (size_t)l * Dc;
        const float* wv = Wv + (size_t)l * Dc * Dc;  const float* bv_ = bv + (size_t)l * Dc;
        const float* wo = Wo + (size_t)l * Dc * Dc;  const float* bo_ = bo + (size_t)l * Dc;
        const float* w1 = W1 + (size_t)l * Dc * Fc;  const float* c1_ = c1 + (size_t)l * Fc;
        const float* w2 = W2 + (size_t)l * Fc * Dc;  const float* c2_ = c2 + (size_t)l * Dc;

        // h = LN1(x) (bf16)
        ln_kernel<1><<<Mc, blk, 0, stream>>>(xin, g1 + (size_t)l * Dc, b1 + (size_t)l * Dc, hbuf, nullptr);
        // q,k,v = h@W + b (bf16)
        gemm_bf16_kernel<EPI_BF16><<<gridD, blk, 0, stream>>>(hbuf, wq, bq_, nullptr, nullptr, qbuf, Mc, Dc, Dc);
        gemm_bf16_kernel<EPI_BF16><<<gridD, blk, 0, stream>>>(hbuf, wk, bk_, nullptr, nullptr, kbuf, Mc, Dc, Dc);
        gemm_bf16_kernel<EPI_BF16><<<gridD, blk, 0, stream>>>(hbuf, wv, bv_, nullptr, nullptr, vbuf, Mc, Dc, Dc);
        // attention
        attn_kernel<<<gridA, blk, 0, stream>>>(qbuf, kbuf, vbuf, mask, obuf);
        // xc = x + o@Wo + bo (fp32)
        gemm_bf16_kernel<EPI_RESID><<<gridD, blk, 0, stream>>>(obuf, wo, bo_, xin, P1, nullptr, Mc, Dc, Dc);
        // h2 = LN2(xc) (bf16)
        ln_kernel<1><<<Mc, blk, 0, stream>>>(P1, g2 + (size_t)l * Dc, b2 + (size_t)l * Dc, hbuf, nullptr);
        // u = gelu(h2@W1 + c1) (bf16)
        gemm_bf16_kernel<EPI_GELU><<<gridF, blk, 0, stream>>>(hbuf, w1, c1_, nullptr, nullptr, ubuf, Mc, Fc, Dc);
        // x' = xc + u@W2 + c2 (fp32)
        gemm_bf16_kernel<EPI_RESID><<<gridD, blk, 0, stream>>>(ubuf, w2, c2_, P1, P0, nullptr, Mc, Dc, Fc);
        xin = P0;
    }
    // final LN -> fp32 output
    ln_kernel<0><<<Mc, blk, 0, stream>>>(xin, gn, bn, nullptr, (float*)d_out);
}
